// Attention_19980187861611
// MI455X (gfx1250) — matrix operand order repaired and checked by reading, not compiled, not run on hardware
//
#include <hip/hip_runtime.h>
#include <hip/hip_bf16.h>
#include <math.h>

// ---------------------------------------------------------------------------
// Tied-row attention (AlphaFold MSA-row style) for gfx1250 / MI455X.
// GEMMs via v_wmma_f32_16x16x32_bf16 (wave32 WMMA), B-tiles staged in LDS
// with global_load_async_to_lds_b128 (ASYNCcnt path). K is compile-time
// (256) and the k-loops are fully unrolled so the scheduler can pipeline
// loads against WMMAs without register rotation.
// ---------------------------------------------------------------------------

#define HEADS   8
#define DIMH    32
#define NB      64      // B
#define NSEQ    256     // N
#define ND      256     // D
#define NINNER  256     // h*dh
#define MROWS   (NB * NSEQ)   // 16384
#define KVW     512
#define KD      256     // GEMM reduction dim (all GEMMs here)

typedef __attribute__((ext_vector_type(16))) __bf16 bf16x16;
typedef __attribute__((ext_vector_type(8)))  float  f32x8;

union BFrag { uint4 u[2]; bf16x16 v; };

__device__ __forceinline__ __bf16 f2bf(float f) {
    union { float f; unsigned u; } x; x.f = f;
    unsigned r = (x.u + 0x7FFFu + ((x.u >> 16) & 1u)) >> 16;
    union { unsigned short s; __bf16 b; } y; y.s = (unsigned short)r;
    return y.b;
}

// A-fragment (16x32 bf16, row-major source, row stride lda elements).
// lane<16: row M=lane, K={0..7,16..23}; lane>=16: row M=lane-16, K={8..15,24..31}
__device__ __forceinline__ bf16x16 load_frag_a(const __bf16* base, int lda, int lane) {
    BFrag f;
    const __bf16* rp = base + (size_t)(lane & 15) * lda + ((lane >> 4) << 3);
    f.u[0] = *(const uint4*)(rp);
    f.u[1] = *(const uint4*)(rp + 16);
    return f.v;
}

// B-fragment (32x16 bf16) from B^T (row-major N x K, row stride ldb).
// Lane holds out-column n=lane%16; lane<16: K={0..7,16..23}; lane>=16: K={8..15,24..31}
__device__ __forceinline__ bf16x16 load_frag_b(const __bf16* baseT, int ldb, int lane) {
    BFrag f;
    const __bf16* rp = baseT + (size_t)(lane & 15) * ldb + ((lane >> 4) << 3);
    f.u[0] = *(const uint4*)(rp);
    f.u[1] = *(const uint4*)(rp + 16);
    return f.v;
}

__device__ __forceinline__ f32x8 wmma_bf16(bf16x16 a, bf16x16 b, f32x8 c) {
    return __builtin_amdgcn_wmma_f32_16x16x32_bf16(
        false, a, false, b, (short)0, c, false, false);
}

// Async copy 16B global -> LDS (CDNA5 ASYNCcnt path).
__device__ __forceinline__ void async_g2l_b128(unsigned lds_off, const void* gptr) {
    unsigned long long ga = (unsigned long long)(uintptr_t)gptr;
    asm volatile("global_load_async_to_lds_b128 %0, %1, off"
                 :: "v"(lds_off), "v"(ga) : "memory");
}
__device__ __forceinline__ void wait_asynccnt0() {
    asm volatile("s_wait_asynccnt 0x0" ::: "memory");
}

// ---------------------------------------------------------------------------
// Elementwise helpers
// ---------------------------------------------------------------------------
__global__ void k_convert_bf16(const float* __restrict__ in, __bf16* __restrict__ out, int n) {
    int i = blockIdx.x * blockDim.x + threadIdx.x;
    if (i < n) out[i] = f2bf(in[i]);
}

// WT[n*K + k] = W[k*N + n]   (f32 -> bf16)
__global__ void k_transpose_w(const float* __restrict__ W, __bf16* __restrict__ WT, int K, int N) {
    int idx = blockIdx.x * blockDim.x + threadIdx.x;
    if (idx >= K * N) return;
    int n = idx / K, k = idx % K;
    WT[idx] = f2bf(W[(size_t)k * N + n]);
}

// xm[i*D + c] = mean over b of x[b, i, c]   (f32 -> bf16)
__global__ void k_mean_x(const float* __restrict__ x, __bf16* __restrict__ xm) {
    int idx = blockIdx.x * blockDim.x + threadIdx.x;   // i*256 + c
    if (idx >= NSEQ * ND) return;
    float acc = 0.f;
#pragma unroll 4
    for (int b = 0; b < NB; ++b) acc += x[(size_t)b * NSEQ * ND + idx];
    xm[idx] = f2bf(acc * (1.0f / NB));
}

// Vt[((rr*8+hh)*32 + d)*256 + j] = KV[rr*256+j, 256 + hh*32 + d]
__global__ void k_vt(const __bf16* __restrict__ KVb, __bf16* __restrict__ Vt) {
    int idx = blockIdx.x * blockDim.x + threadIdx.x;
    if (idx >= NB * HEADS * DIMH * NSEQ) return;
    int j  = idx & 255;
    int d  = (idx >> 8) & 31;
    int hh = (idx >> 13) & 7;
    int rr = idx >> 16;
    Vt[idx] = KVb[((size_t)rr * NSEQ + j) * KVW + NINNER + hh * DIMH + d];
}

// ---------------------------------------------------------------------------
// Staged bf16 WMMA GEMM: C[M,N] = alpha*(A @ B) (+bias) (sigmoid), K = KD.
// A row-major (lda=KD), BT row-major N x KD.
// Block = 256 threads (8 waves): 128-row M-tile x 64-col N-quad.
// The 64xKD B^T quad (32 KB, contiguous) is async-copied to LDS once per
// block and shared by all 8 waves. The k-loop is fully unrolled: 8 global
// A-fragment loads + 64 ds_load_b128 + 32 WMMAs in one schedulable block.
// mode: 1=add bias, 2=sigmoid, 4=write f32, 8=write bf16
// ---------------------------------------------------------------------------
__global__ void k_gemm(const __bf16* __restrict__ A, const __bf16* __restrict__ BT,
                       const float* __restrict__ bias,
                       float* __restrict__ outF, __bf16* __restrict__ outB,
                       int M, int N, float alpha, int mode) {
    __shared__ __align__(16) __bf16 sB[64 * KD];   // 32 KB

    int tid  = threadIdx.x;
    int lane = tid & 31;
    int wv   = tid >> 5;                 // wave 0..7
    int nquads = N >> 6;
    int mt = blockIdx.x / nquads;        // 128-row tile index
    int nq = blockIdx.x % nquads;

    // ---- async-stage the B^T quad: 64 rows x KD bf16, contiguous ----
    {
        const __bf16* qbase = BT + (size_t)nq * 64 * KD;
#pragma unroll
        for (int i = tid; i < (64 * KD) >> 3; i += 256) {
            unsigned loff = (unsigned)(uintptr_t)(&sB[(size_t)i << 3]);
            async_g2l_b128(loff, qbase + ((size_t)i << 3));
        }
        wait_asynccnt0();
        __syncthreads();
    }

    int m0 = mt * 128 + wv * 16;
    const __bf16* Abase = A + (size_t)m0 * KD;

    f32x8 c[4];
#pragma unroll
    for (int t = 0; t < 4; ++t) c[t] = (f32x8){0,0,0,0,0,0,0,0};

#pragma unroll
    for (int kk = 0; kk < KD; kk += 32) {
        bf16x16 a = load_frag_a(Abase + kk, KD, lane);
#pragma unroll
        for (int t = 0; t < 4; ++t) {
            bf16x16 b = load_frag_b(&sB[(size_t)(t * 16) * KD + kk], KD, lane);
            c[t] = wmma_bf16(a, b, c[t]);
        }
    }

    int half = lane >> 4;
#pragma unroll
    for (int t = 0; t < 4; ++t) {
        int n = nq * 64 + t * 16 + (lane & 15);
        float bv = (mode & 1) ? bias[n] : 0.0f;
#pragma unroll
        for (int v = 0; v < 8; ++v) {
            int row = m0 + v + half * 8;
            float val = alpha * c[t][v] + bv;
            if (mode & 2) val = 1.0f / (1.0f + __expf(-val));
            size_t o = (size_t)row * N + n;
            if (mode & 4) outF[o] = val;
            if (mode & 8) outB[o] = f2bf(val);
        }
    }
}

// ---------------------------------------------------------------------------
// Fused dots + bias + mask + softmax.
// One wave per (rr, hh, 16-row i-tile). K-dim = dh = 32 -> 1 wmma per
// 16-column tile; 16 tiles cover the full j=256 row held in registers.
// ---------------------------------------------------------------------------
__global__ void k_dots_softmax(const __bf16* __restrict__ qt,
                               const __bf16* __restrict__ KVb,
                               const float* __restrict__ attn_bias,
                               const int* __restrict__ mask,
                               __bf16* __restrict__ attn) {
    int lane = threadIdx.x & 31;
    int i0 = blockIdx.x * 16;
    int hh = blockIdx.y;
    int rr = blockIdx.z;
    int jlane = lane & 15;
    int half = lane >> 4;

    bf16x16 a = load_frag_a(qt + (size_t)i0 * NINNER + hh * DIMH, NINNER, lane);

    f32x8 c[16];
#pragma unroll
    for (int t = 0; t < 16; ++t) {
        c[t] = (f32x8){0,0,0,0,0,0,0,0};
        bf16x16 b = load_frag_b(KVb + ((size_t)rr * NSEQ + t * 16) * KVW + hh * DIMH,
                                KVW, lane);
        c[t] = wmma_bf16(a, b, c[t]);
    }

    // bias + pair mask
    int mr[8];
#pragma unroll
    for (int v = 0; v < 8; ++v) mr[v] = mask[rr * NSEQ + i0 + v + half * 8];
#pragma unroll
    for (int t = 0; t < 16; ++t) {
        int j = t * 16 + jlane;
        int mc = mask[rr * NSEQ + j];
#pragma unroll
        for (int v = 0; v < 8; ++v) {
            int i = i0 + v + half * 8;
            float val = c[t][v] + attn_bias[((size_t)hh * NSEQ + i) * NSEQ + j];
            c[t][v] = (mr[v] && mc) ? val : -3.402823466e38f;
        }
    }

    // softmax over j (16 in-lane tiles x 16 lanes per half-wave)
#pragma unroll
    for (int v = 0; v < 8; ++v) {
        float m = -3.402823466e38f;
#pragma unroll
        for (int t = 0; t < 16; ++t) m = fmaxf(m, c[t][v]);
        m = fmaxf(m, __shfl_xor(m, 1));
        m = fmaxf(m, __shfl_xor(m, 2));
        m = fmaxf(m, __shfl_xor(m, 4));
        m = fmaxf(m, __shfl_xor(m, 8));
        float s = 0.f;
#pragma unroll
        for (int t = 0; t < 16; ++t) {
            float e = __expf(c[t][v] - m);
            c[t][v] = e;
            s += e;
        }
        s += __shfl_xor(s, 1);
        s += __shfl_xor(s, 2);
        s += __shfl_xor(s, 4);
        s += __shfl_xor(s, 8);
        float inv = 1.0f / s;
        int i = i0 + v + half * 8;
        size_t rowbase = ((size_t)(rr * HEADS + hh) * NSEQ + i) * NSEQ;
#pragma unroll
        for (int t = 0; t < 16; ++t)
            attn[rowbase + t * 16 + jlane] = f2bf(c[t][v] * inv);
    }
}

// ---------------------------------------------------------------------------
// out_head = attn @ V fused with sigmoid gating; writes bf16 outg laid out
// as (B, N, h*dh) row-major (ready for the output projection GEMM).
// ---------------------------------------------------------------------------
__global__ void k_attn_v(const __bf16* __restrict__ attn,
                         const __bf16* __restrict__ Vt,
                         const float* __restrict__ G,
                         __bf16* __restrict__ outg) {
    int lane = threadIdx.x & 31;
    int i0 = blockIdx.x * 16;
    int hh = blockIdx.y;
    int rr = blockIdx.z;
    int jlane = lane & 15;
    int half = lane >> 4;

    const __bf16* Abase = attn + ((size_t)(rr * HEADS + hh) * NSEQ + i0) * NSEQ;
    const __bf16* Vbase = Vt + (size_t)(rr * HEADS + hh) * DIMH * NSEQ;

    f32x8 c0 = (f32x8){0,0,0,0,0,0,0,0};
    f32x8 c1 = (f32x8){0,0,0,0,0,0,0,0};
#pragma unroll
    for (int kk = 0; kk < NSEQ; kk += 32) {
        bf16x16 a  = load_frag_a(Abase + kk, NSEQ, lane);
        bf16x16 b0 = load_frag_b(Vbase + kk, NSEQ, lane);
        bf16x16 b1 = load_frag_b(Vbase + (size_t)16 * NSEQ + kk, NSEQ, lane);
        c0 = wmma_bf16(a, b0, c0);
        c1 = wmma_bf16(a, b1, c1);
    }

#pragma unroll
    for (int dt = 0; dt < 2; ++dt) {
        int d = dt * 16 + jlane;
#pragma unroll
        for (int v = 0; v < 8; ++v) {
            int i = i0 + v + half * 8;
            size_t idx = ((size_t)rr * NSEQ + i) * NINNER + hh * DIMH + d;
            float val = (dt == 0 ? c0[v] : c1[v]) * G[idx];
            outg[idx] = f2bf(val);
        }
    }
}

// ---------------------------------------------------------------------------
// Host orchestration
// ---------------------------------------------------------------------------
extern "C" void kernel_launch(void* const* d_in, const int* in_sizes, int n_in,
                              void* d_out, int out_size, void* d_ws, size_t ws_size,
                              hipStream_t stream) {
    (void)in_sizes; (void)n_in; (void)out_size; (void)ws_size;

    const float* x        = (const float*)d_in[0];
    const int*   mask     = (const int*)d_in[1];
    const float* attnBias = (const float*)d_in[2];
    const float* Wq       = (const float*)d_in[3];
    const float* Wkv      = (const float*)d_in[4];
    const float* Wg       = (const float*)d_in[5];
    const float* bg       = (const float*)d_in[6];
    const float* Wo       = (const float*)d_in[7];
    const float* bo       = (const float*)d_in[8];
    float*       out      = (float*)d_out;

    char* w = (char*)d_ws;
    size_t o = 0;
    __bf16* xb   = (__bf16*)(w + o); o += (size_t)MROWS * ND * 2;        // 8 MB
    __bf16* WqT  = (__bf16*)(w + o); o += (size_t)ND * NINNER * 2;       // 128 KB
    __bf16* WkvT = (__bf16*)(w + o); o += (size_t)ND * KVW * 2;          // 256 KB
    __bf16* WgT  = (__bf16*)(w + o); o += (size_t)ND * NINNER * 2;
    __bf16* WoT  = (__bf16*)(w + o); o += (size_t)NINNER * ND * 2;
    __bf16* xm   = (__bf16*)(w + o); o += (size_t)NSEQ * ND * 2;
    __bf16* qt   = (__bf16*)(w + o); o += (size_t)NSEQ * NINNER * 2;
    __bf16* KVb  = (__bf16*)(w + o); o += (size_t)MROWS * KVW * 2;       // 16 MB
    float*  G    = (float*) (w + o); o += (size_t)MROWS * NINNER * 4;    // 16 MB
    __bf16* Vt   = (__bf16*)(w + o); o += (size_t)NB * HEADS * DIMH * NSEQ * 2; // 8 MB
    __bf16* attn = (__bf16*)(w + o); o += (size_t)NB * HEADS * NSEQ * NSEQ * 2; // 64 MB
    __bf16* outg = (__bf16*)(w + o); o += (size_t)MROWS * NINNER * 2;    // 8 MB

    const float scale = 0.17677669529663687f;   // dh^-0.5

    // 1) conversions / transposes
    k_convert_bf16<<<(MROWS * ND) / 256, 256, 0, stream>>>(x, xb, MROWS * ND);
    k_transpose_w<<<(ND * NINNER) / 256, 256, 0, stream>>>(Wq, WqT, ND, NINNER);
    k_transpose_w<<<(ND * KVW) / 256, 256, 0, stream>>>(Wkv, WkvT, ND, KVW);
    k_transpose_w<<<(ND * NINNER) / 256, 256, 0, stream>>>(Wg, WgT, ND, NINNER);
    k_transpose_w<<<(NINNER * ND) / 256, 256, 0, stream>>>(Wo, WoT, NINNER, ND);
    k_mean_x<<<(NSEQ * ND) / 256, 256, 0, stream>>>(x, xm);

    // 2) projections (staged WMMA GEMMs; grid = (M/128)*(N/64) blocks of 256)
    // qt = (mean_b x) @ Wq * scale        -> bf16
    k_gemm<<<(NSEQ / 128) * (NINNER / 64), 256, 0, stream>>>(
        xm, WqT, nullptr, nullptr, qt, NSEQ, NINNER, scale, 8);
    // KV = x @ Wkv                        -> bf16
    k_gemm<<<(MROWS / 128) * (KVW / 64), 256, 0, stream>>>(
        xb, WkvT, nullptr, nullptr, KVb, MROWS, KVW, 1.0f, 8);
    // G = sigmoid(x @ Wg + bg)            -> f32
    k_gemm<<<(MROWS / 128) * (NINNER / 64), 256, 0, stream>>>(
        xb, WgT, bg, G, nullptr, MROWS, NINNER, 1.0f, 1 | 2 | 4);

    // 3) V transpose for contiguous B-fragments
    k_vt<<<(NB * HEADS * DIMH * NSEQ) / 256, 256, 0, stream>>>(KVb, Vt);

    // 4) fused dots + bias + mask + softmax
    k_dots_softmax<<<dim3(NSEQ / 16, HEADS, NB), 32, 0, stream>>>(
        qt, KVb, attnBias, mask, attn);

    // 5) attn @ V fused with gating
    k_attn_v<<<dim3(NSEQ / 16, HEADS, NB), 32, 0, stream>>>(attn, Vt, G, outg);

    // 6) output projection: out = outg @ Wo + bo  -> f32
    k_gemm<<<(MROWS / 128) * (ND / 64), 256, 0, stream>>>(
        outg, WoT, bo, out, nullptr, MROWS, ND, 1.0f, 1 | 4);
}
